// VoxelNeXtHead_85091892069086
// MI455X (gfx1250) — compile-verified, matrix operand order checked by reading, not run on hardware
//
#include <hip/hip_runtime.h>
#include <stdint.h>

#define N_PTS 150000
#define C     128
#define KK    9
#define NH    5
#define MAXO  3
#define EPSV  1e-5f

#define MT          2            // 16-row M-tiles per block
#define PTS_PER_BLK (MT * 16)    // 32 points per block

typedef __attribute__((ext_vector_type(16))) __bf16 v16bf;
typedef __attribute__((ext_vector_type(8)))  float  v8f;

// LDS row pitch in bf16 elements: 9*128 data + 16 pad (32B) to stagger banks.
#define LDS_PITCH (KK * C + 16)   // 1168 bf16 = 2336 B per point-row

// ---------------------------------------------------------------- utilities
__global__ void zero_stats_kernel(float* stats) {
    int i = blockIdx.x * 256 + threadIdx.x;
    if (i < NH * C * 2) stats[i] = 0.0f;
}

// W1 fp32 [h][k][c][d]  ->  W1t bf16 [h][k][d][c]   (so B-frag loads are contiguous)
__global__ void pack_w1_kernel(const float* __restrict__ W1, __bf16* __restrict__ W1t) {
    int idx = blockIdx.x * 256 + threadIdx.x;         // enumerates output layout
    if (idx >= NH * KK * C * C) return;
    int c  = idx & (C - 1);
    int d  = (idx >> 7) & (C - 1);
    int hk = idx >> 14;
    float v = W1[((size_t)hk * C + c) * C + d];
    W1t[idx] = (__bf16)v;
}

// ------------------------------------------------------ main gather + WMMA
__global__ __launch_bounds__(256)
void gconv_wmma_kernel(const float* __restrict__ x,
                       const int*   __restrict__ nbr,
                       const __bf16* __restrict__ W1t,
                       float* __restrict__ acc,        // [NH][N][C] fp32
                       float* __restrict__ stats)      // [2][NH][C]: sum, sumsq
{
    __shared__ __attribute__((aligned(32))) __bf16 lds[PTS_PER_BLK * LDS_PITCH];

    const int tid = threadIdx.x;
    const int n0  = blockIdx.x * PTS_PER_BLK;         // last block: second tile OOB

    // ---- stage 32 points x 9 neighbors x 128 ch as bf16 into LDS (float4 gathers)
    for (int e = tid; e < PTS_PER_BLK * KK * (C / 4); e += 256) {
        int c4 = e & 31;
        int km = e >> 5;
        int k  = km % KK;
        int m  = km / KK;
        int n  = n0 + m;
        float4 v = make_float4(0.f, 0.f, 0.f, 0.f);
        if (n < N_PTS) {
            int idx = nbr[(size_t)n * KK + k];
            if ((unsigned)idx < (unsigned)N_PTS)
                v = *(const float4*)(x + (size_t)idx * C + c4 * 4);
        }
        __bf16* dst = &lds[m * LDS_PITCH + k * C + c4 * 4];
        dst[0] = (__bf16)v.x; dst[1] = (__bf16)v.y;
        dst[2] = (__bf16)v.z; dst[3] = (__bf16)v.w;
    }
    __syncthreads();

    const int wave = tid >> 5;         // 0..7 -> d-tile (wave*16 .. wave*16+15)
    const int lane = tid & 31;
    const int nL   = lane & 15;
    const int half = lane >> 4;

    v8f accv[MT][NH];
    #pragma unroll
    for (int mt = 0; mt < MT; ++mt)
        #pragma unroll
        for (int h = 0; h < NH; ++h) accv[mt][h] = (v8f)(0.0f);

    // A frag: lane holds row m=nL, K = 16*half .. 16*half+15 (contiguous k order, same as B)
    const __bf16* aBase = &lds[nL * LDS_PITCH + 16 * half];
    // B frag: lane holds column d = wave*16+nL, K = 16*half .. +15 (contiguous in W1t)
    const size_t bLane = (size_t)(wave * 16 + nL) * C + 16 * half;

    for (int k = 0; k < KK; ++k) {
        #pragma unroll
        for (int cb = 0; cb < 4; ++cb) {             // c_base = cb*32, K-step 32
            v16bf af[MT];
            #pragma unroll
            for (int mt = 0; mt < MT; ++mt)
                af[mt] = *(const v16bf*)(aBase + mt * 16 * LDS_PITCH + k * C + cb * 32);
            #pragma unroll
            for (int h = 0; h < NH; ++h) {
                const __bf16* bp = W1t + (size_t)(h * KK + k) * C * C + bLane + cb * 32;
                v16bf bfrag = *(const v16bf*)bp;     // one B load feeds MT WMMAs
                #pragma unroll
                for (int mt = 0; mt < MT; ++mt)
                    accv[mt][h] = __builtin_amdgcn_wmma_f32_16x16x32_bf16(
                        /*neg_a=*/false, af[mt], /*neg_b=*/false, bfrag,
                        /*c_mod=*/(short)0, accv[mt][h],
                        /*reuse_a=*/false, /*reuse_b=*/false);
            }
        }
    }

    // ---- store acc (fp32) + block-partial sum/sumsq per (h,d) via atomics
    const int d = wave * 16 + nL;
    #pragma unroll
    for (int mt = 0; mt < MT; ++mt) {
        const int nbase = n0 + mt * 16;
        if (nbase >= N_PTS) break;                   // whole tile OOB (N % 16 == 0)
        #pragma unroll
        for (int h = 0; h < NH; ++h) {
            float s = 0.f, s2 = 0.f;
            #pragma unroll
            for (int j = 0; j < 8; ++j) {            // VGPR j -> row m = j + 8*half
                float v = accv[mt][h][j];
                int m = j + 8 * half;
                acc[((size_t)h * N_PTS + nbase + m) * C + d] = v;
                s  += v;
                s2 += v * v;
            }
            s  += __shfl_xor(s, 16, 32);             // fold two half-waves (wave32)
            s2 += __shfl_xor(s2, 16, 32);
            if (half == 0) {
                atomicAdd(&stats[h * C + d], s);
                atomicAdd(&stats[NH * C + h * C + d], s2);
            }
        }
    }
}

// ------------------------------------------------------ stats -> scale/shift
__global__ void finalize_stats_kernel(const float* __restrict__ stats,
                                      const float* __restrict__ gamma,
                                      const float* __restrict__ beta,
                                      float* __restrict__ ss) {   // [2][NH][C]
    int i = blockIdx.x * 256 + threadIdx.x;
    if (i >= NH * C) return;
    float inv_n = 1.0f / (float)N_PTS;
    float mean = stats[i] * inv_n;
    float var  = stats[NH * C + i] * inv_n - mean * mean;
    float rstd = rsqrtf(var + EPSV);
    float sc   = rstd * gamma[i];
    ss[i]          = sc;                  // scale
    ss[NH * C + i] = beta[i] - mean * sc; // shift
}

// ------------------------------------------------------ normalize+relu+W2
__global__ __launch_bounds__(256)
void head_out_kernel(const float* __restrict__ acc,
                     const float* __restrict__ ss,
                     const float* __restrict__ W2,   // [NH][C][MAXO]
                     const float* __restrict__ b2,   // [NH][MAXO]
                     float* __restrict__ out)        // [NH][N][MAXO]
{
    const int lane = threadIdx.x & 31;
    const int wave = threadIdx.x >> 5;
    long long row = (long long)blockIdx.x * 8 + wave;   // row = h*N + n
    if (row >= (long long)NH * N_PTS) return;
    int h = (int)(row / N_PTS);

    const float4 a4 = ((const float4*)(acc + (size_t)row * C))[lane];
    float av[4] = {a4.x, a4.y, a4.z, a4.w};

    float p0 = 0.f, p1 = 0.f, p2 = 0.f;
    #pragma unroll
    for (int i = 0; i < 4; ++i) {
        int c = lane * 4 + i;
        float v = av[i] * ss[h * C + c] + ss[NH * C + h * C + c];
        v = v > 0.f ? v : 0.f;
        const float* w = &W2[(size_t)(h * C + c) * MAXO];
        p0 += v * w[0]; p1 += v * w[1]; p2 += v * w[2];
    }
    #pragma unroll
    for (int off = 16; off > 0; off >>= 1) {
        p0 += __shfl_xor(p0, off, 32);
        p1 += __shfl_xor(p1, off, 32);
        p2 += __shfl_xor(p2, off, 32);
    }
    if (lane == 0) {
        const int OC[NH] = {3, 2, 1, 3, 2};
        float* o = out + (size_t)row * MAXO;
        o[0] = (0 < OC[h]) ? (p0 + b2[h * MAXO + 0]) : 0.0f;
        o[1] = (1 < OC[h]) ? (p1 + b2[h * MAXO + 1]) : 0.0f;
        o[2] = (2 < OC[h]) ? (p2 + b2[h * MAXO + 2]) : 0.0f;
    }
}

// ---------------------------------------------------------------- launcher
extern "C" void kernel_launch(void* const* d_in, const int* in_sizes, int n_in,
                              void* d_out, int out_size, void* d_ws, size_t ws_size,
                              hipStream_t stream) {
    const float* x     = (const float*)d_in[0];   // (N, C)
    const int*   nbr   = (const int*)  d_in[1];   // (N, KK)
    const float* W1    = (const float*)d_in[2];   // (NH, KK, C, C)
    const float* gamma = (const float*)d_in[3];   // (NH, C)
    const float* beta  = (const float*)d_in[4];   // (NH, C)
    const float* W2    = (const float*)d_in[5];   // (NH, C, MAXO)
    const float* b2    = (const float*)d_in[6];   // (NH, MAXO)
    float* out = (float*)d_out;                   // (NH, N, MAXO)

    // workspace layout (256B-aligned sections)
    char* ws = (char*)d_ws;
    size_t acc_bytes = (size_t)NH * N_PTS * C * sizeof(float);      // 384 MB
    float*  acc   = (float*)ws;
    float*  stats = (float*)(ws + acc_bytes);                        // 2*NH*C f32
    float*  ss    = (float*)(ws + acc_bytes + 8192);                 // 2*NH*C f32
    __bf16* W1t   = (__bf16*)(ws + acc_bytes + 16384);               // NH*KK*C*C bf16

    zero_stats_kernel<<<(NH * C * 2 + 255) / 256, 256, 0, stream>>>(stats);
    pack_w1_kernel<<<(NH * KK * C * C + 255) / 256, 256, 0, stream>>>(W1, W1t);
    gconv_wmma_kernel<<<(N_PTS + PTS_PER_BLK - 1) / PTS_PER_BLK, 256, 0, stream>>>(
        x, nbr, W1t, acc, stats);
    finalize_stats_kernel<<<(NH * C + 255) / 256, 256, 0, stream>>>(stats, gamma, beta, ss);
    head_out_kernel<<<(int)(((long long)NH * N_PTS + 7) / 8), 256, 0, stream>>>(acc, ss, W2, b2, out);
}
